// QuantMoeBlock_38689065402897
// MI455X (gfx1250) — matrix operand order repaired and checked by reading, not compiled, not run on hardware
//
#include <hip/hip_runtime.h>
#include <hip/hip_bf16.h>
#include <cstdint>

// ---------------- problem constants ----------------
#define T_TOK 8192   // B*S
#define HDIM  2048
#define IDIM  1408
#define SIDIM 5632
#define NEXP  16

// ---------------- GEMM tiling ----------------
#define BM 128
#define BN 128
#define BK 64
#define LDK 72       // LDS row stride (ushorts) = 144B: TDM pad reproduces this
#define TILE_USH (BM * LDK)   // 9216 ushorts = 18432 B per tile buffer

typedef __attribute__((ext_vector_type(16))) __bf16 v16bf;
typedef __attribute__((ext_vector_type(8)))  float  v8f;

typedef unsigned int tdm_u32x4 __attribute__((ext_vector_type(4)));
typedef int          tdm_i32x8 __attribute__((ext_vector_type(8)));
typedef int          tdm_i32x4 __attribute__((ext_vector_type(4)));

union Frag {
  v16bf v;
  uint4 q[2];
};

__device__ __forceinline__ unsigned int f2bf(float f) {
  union { float f; unsigned int u; } cv;
  cv.f = f;
  unsigned int u = cv.u;
  unsigned int r = u + 0x7FFFu + ((u >> 16) & 1u);  // round-to-nearest-even
  return r >> 16;
}

// ---- TDM: async-load one BM x BK bf16 tile (row stride K elems) into LDS ----
// LDS layout produced: row stride 144B (128B data + 16B pad), matching LDK.
__device__ __forceinline__ void tdm_load_tile(unsigned int lds_byte,
                                              const unsigned short* gbase,
                                              int K) {
  unsigned long long ga = (unsigned long long)(uintptr_t)gbase;
  tdm_u32x4 g0;
  g0[0] = 1u;                                        // count=1, user descriptor
  g0[1] = lds_byte;                                  // lds_addr
  g0[2] = (unsigned int)(ga & 0xFFFFFFFFu);          // global_addr[31:0]
  g0[3] = (unsigned int)((ga >> 32) & 0x01FFFFFFu)   // global_addr[56:32]
          | (2u << 30);                              // type=2 (image)
  unsigned int td0 = (unsigned int)K;                // tensor_dim0 (elems)
  unsigned int td1 = 0x40000000u;                    // tensor_dim1 (huge)
  unsigned long long s0 = (unsigned long long)K;     // tensor_dim0_stride
  tdm_i32x8 g1;
  g1[0] = (int)((1u << 16)        // data_size = 2 bytes (bf16)
                | (1u << 20)      // pad_enable
                | (4u << 22)      // pad_interval: 32 DWORDs (=128B)
                | (3u << 25));    // pad_amount:   4 DWORDs (=16B)
  g1[1] = (int)((td0 & 0xFFFFu) << 16);
  g1[2] = (int)((td0 >> 16) | ((td1 & 0xFFFFu) << 16));
  g1[3] = (int)((td1 >> 16) | ((unsigned)BK << 16)); // tile_dim0 = 64
  g1[4] = (int)((unsigned)BM);                       // tile_dim1 = 128, tile_dim2=0
  g1[5] = (int)(s0 & 0xFFFFFFFFu);
  g1[6] = (int)((s0 >> 32) & 0xFFFFu);               // dim1_stride low16 = 0
  g1[7] = 0;
  tdm_i32x4 g2 = {1, 0, 0, 0};                       // tensor_dim2=1
  tdm_i32x4 g3 = {0, 0, 0, 0};
  tdm_i32x8 g4 = {0, 0, 0, 0, 0, 0, 0, 0};           // extra operand (clang-23 form)
  __builtin_amdgcn_tensor_load_to_lds(g0, g1, g2, g3, g4, 0);
}

// ---------------- fp32 -> bf16 cast (activations and staged weights) ----------------
__global__ __launch_bounds__(256) void cast_bf16_kernel(
    const float* __restrict__ x, unsigned short* __restrict__ xb) {
  size_t i = (size_t)blockIdx.x * 256 + threadIdx.x;   // handles 8 floats
  const float4* src = (const float4*)x;
  float4 f0 = src[i * 2 + 0];
  float4 f1 = src[i * 2 + 1];
  uint4 o;
  o.x = f2bf(f0.x) | (f2bf(f0.y) << 16);
  o.y = f2bf(f0.z) | (f2bf(f0.w) << 16);
  o.z = f2bf(f1.x) | (f2bf(f1.y) << 16);
  o.w = f2bf(f1.z) | (f2bf(f1.w) << 16);
  ((uint4*)xb)[i] = o;
}

// ---------------- router: logits, softmax, top-2, shared gate ----------------
__global__ __launch_bounds__(256) void router_kernel(
    const float* __restrict__ x, const float* __restrict__ rw,
    const float* __restrict__ sgw, float* __restrict__ logits_out,
    float* __restrict__ combine, float* __restrict__ sgate) {
  const int lane = threadIdx.x & 31;
  const int wave = threadIdx.x >> 5;
  const int t = blockIdx.x * 8 + wave;           // one token per wave32

  const float* xrow = x + (size_t)t * HDIM;
  float xr[64];
#pragma unroll
  for (int j = 0; j < 64; ++j) xr[j] = xrow[j * 32 + lane];

  float s[17];
  for (int e = 0; e < 17; ++e) {
    const float* w = (e < 16) ? (rw + (size_t)e * HDIM) : sgw;
    float p = 0.0f;
#pragma unroll
    for (int j = 0; j < 64; ++j) p += xr[j] * w[j * 32 + lane];
#pragma unroll
    for (int m = 16; m > 0; m >>= 1) p += __shfl_xor(p, m, 32);
    s[e] = p;
  }

  float mx = s[0];
#pragma unroll
  for (int e = 1; e < 16; ++e) mx = fmaxf(mx, s[e]);
  float pr[16], sum = 0.0f;
#pragma unroll
  for (int e = 0; e < 16; ++e) { pr[e] = __expf(s[e] - mx); sum += pr[e]; }
  float inv = 1.0f / sum;
#pragma unroll
  for (int e = 0; e < 16; ++e) pr[e] *= inv;

  int i1 = 0;
#pragma unroll
  for (int e = 1; e < 16; ++e) if (pr[e] > pr[i1]) i1 = e;
  int i2 = (i1 == 0) ? 1 : 0;
#pragma unroll
  for (int e = 0; e < 16; ++e) if (e != i1 && pr[e] > pr[i2]) i2 = e;

  if (lane < 16) {
    logits_out[(size_t)t * 16 + lane] = s[lane];
    combine[(size_t)t * 16 + lane] =
        (lane == i1 || lane == i2) ? pr[lane] : 0.0f;
  }
  if (lane == 0) sgate[t] = 1.0f / (1.0f + __expf(-s[16]));
}

// ---------------- deterministic token dispatch (one block per expert) ----------------
__global__ __launch_bounds__(256) void dispatch_kernel(
    const float* __restrict__ combine, int* __restrict__ counts,
    int* __restrict__ idxlist) {
  const int e = blockIdx.x;
  const int tid = threadIdx.x;
  const int lane = tid & 31;
  const int wave = tid >> 5;
  __shared__ int wsum[8];
  __shared__ int base;
  if (tid == 0) base = 0;
  __syncthreads();
  for (int chunk = 0; chunk < T_TOK; chunk += 256) {
    int t = chunk + tid;
    bool sel = combine[(size_t)t * NEXP + e] > 0.0f;
    unsigned int m = (unsigned int)__ballot(sel);  // wave32: low 32 bits
    int lpre = __popc(m & ((1u << lane) - 1u));
    if (lane == 0) wsum[wave] = __popc(m);
    __syncthreads();
    int wbase = 0;
    for (int w = 0; w < wave; ++w) wbase += wsum[w];
    if (sel) idxlist[e * T_TOK + base + wbase + lpre] = t;
    __syncthreads();
    if (tid == 0) {
      int tot = 0;
      for (int w = 0; w < 8; ++w) tot += wsum[w];
      base += tot;
    }
    __syncthreads();
  }
  if (tid == 0) counts[e] = base;
  if (e == 0 && tid == 0) counts[NEXP] = T_TOK;   // "count" for shared expert
}

// ---------------- fused gate/up GEMM + SwiGLU ----------------
// A tile gathered via VGPR path; G/U weight tiles via TDM (double-buffered).
__global__ __launch_bounds__(256) void moe_gateup_kernel(
    const unsigned short* __restrict__ xb, const unsigned short* __restrict__ gwb,
    const unsigned short* __restrict__ uwb, unsigned short* __restrict__ hbuf,
    const int* __restrict__ idx, const int* __restrict__ cnt, int K, int N) {
  const int count = *cnt;
  const int tm = blockIdx.x;
  const int tn = blockIdx.y;
  if (tm * BM >= count) return;

  // [sA | sG0 | sG1 | sU0 | sU1]
  __shared__ unsigned short smem[5 * TILE_USH];

  const int tid = threadIdx.x;
  const int lane = tid & 31;
  const int wave = tid >> 5;
  const int wm = wave >> 2;
  const int wn = wave & 3;
  const int half = lane >> 4;
  const int mn = lane & 15;
  const int c8 = tid & 7;

  int rowL[4], tokA[4];
#pragma unroll
  for (int t = 0; t < 4; ++t) {
    int row = (tid + t * 256) >> 3;
    rowL[t] = row;
    int g = tm * BM + row;
    tokA[t] = idx ? ((g < count) ? idx[g] : 0) : g;
  }

  v8f cg[4][2], cu[4][2];
  v8f vzero = {};
#pragma unroll
  for (int mi = 0; mi < 4; ++mi)
#pragma unroll
    for (int ni = 0; ni < 2; ++ni) { cg[mi][ni] = vzero; cu[mi][ni] = vzero; }

  const unsigned short* gBase = gwb + (size_t)(tn * BN) * K;
  const unsigned short* uBase = uwb + (size_t)(tn * BN) * K;
  const int nChunks = K / BK;

  uint4 ra[4];
#pragma unroll
  for (int t = 0; t < 4; ++t)
    ra[t] = *(const uint4*)(xb + (size_t)tokA[t] * K + c8 * 8);
  if (wave == 0) {
    tdm_load_tile((1 * TILE_USH) * 2, gBase, K);
    tdm_load_tile((3 * TILE_USH) * 2, uBase, K);
  }

  for (int i = 0; i < nChunks; ++i) {
    const int buf = i & 1;
    const int gOff = (1 + buf) * TILE_USH;
    const int uOff = (3 + buf) * TILE_USH;
    __syncthreads();                       // prev compute done: sA reusable
#pragma unroll
    for (int t = 0; t < 4; ++t)
      *(uint4*)&smem[rowL[t] * LDK + c8 * 8] = ra[t];
    if (wave == 0) __builtin_amdgcn_s_wait_tensorcnt(0);  // G/U chunk i landed
    __syncthreads();                       // publish sA stores + TDM tiles
    if (i + 1 < nChunks) {
      const int kn = (i + 1) * BK;
#pragma unroll
      for (int t = 0; t < 4; ++t)
        ra[t] = *(const uint4*)(xb + (size_t)tokA[t] * K + kn + c8 * 8);
      if (wave == 0) {
        tdm_load_tile((1 + (buf ^ 1)) * TILE_USH * 2, gBase + kn, K);
        tdm_load_tile((3 + (buf ^ 1)) * TILE_USH * 2, uBase + kn, K);
      }
    }

#pragma unroll
    for (int ks = 0; ks < BK; ks += 32) {
      Frag a[4], bg[2], bu[2];
#pragma unroll
      for (int mi = 0; mi < 4; ++mi) {
        int base = (wm * 64 + mi * 16 + mn) * LDK + ks + half * 8;
        a[mi].q[0] = *(const uint4*)&smem[base];
        a[mi].q[1] = *(const uint4*)&smem[base + 16];
      }
#pragma unroll
      for (int ni = 0; ni < 2; ++ni) {
        int base = (wn * 32 + ni * 16 + mn) * LDK + ks + half * 8;
        bg[ni].q[0] = *(const uint4*)&smem[gOff + base];
        bg[ni].q[1] = *(const uint4*)&smem[gOff + base + 16];
        bu[ni].q[0] = *(const uint4*)&smem[uOff + base];
        bu[ni].q[1] = *(const uint4*)&smem[uOff + base + 16];
      }
#pragma unroll
      for (int mi = 0; mi < 4; ++mi)
#pragma unroll
        for (int ni = 0; ni < 2; ++ni) {
          cg[mi][ni] = __builtin_amdgcn_wmma_f32_16x16x32_bf16(
              false, a[mi].v, false, bg[ni].v, (short)0, cg[mi][ni], false, false);
          cu[mi][ni] = __builtin_amdgcn_wmma_f32_16x16x32_bf16(
              false, a[mi].v, false, bu[ni].v, (short)0, cu[mi][ni], false, false);
        }
    }
  }

#pragma unroll
  for (int mi = 0; mi < 4; ++mi)
#pragma unroll
    for (int ni = 0; ni < 2; ++ni)
#pragma unroll
      for (int r = 0; r < 8; ++r) {
        float g = cg[mi][ni][r];
        float u = cu[mi][ni][r];
        float h = (g / (1.0f + __expf(-g))) * u;
        int p = tm * BM + wm * 64 + mi * 16 + half * 8 + r;   // compacted row
        int icol = tn * BN + wn * 32 + ni * 16 + mn;
        hbuf[(size_t)p * N + icol] = (unsigned short)f2bf(h);
      }
}

// ---------------- down projection + per-token combine, scatter by token ----------------
// Both tiles contiguous -> pure TDM double-buffered pipeline.
__global__ __launch_bounds__(256) void moe_down_kernel(
    const unsigned short* __restrict__ hb, const unsigned short* __restrict__ dwb,
    const float* __restrict__ cePtr, int ceStride, const int* __restrict__ idx,
    const int* __restrict__ cnt, float* __restrict__ out, int K, int beta) {
  const int count = *cnt;
  const int tm = blockIdx.x;
  const int tn = blockIdx.y;
  if (tm * BM >= count) return;

  // [sA0 | sA1 | sB0 | sB1]
  __shared__ unsigned short smem[4 * TILE_USH];

  const int tid = threadIdx.x;
  const int lane = tid & 31;
  const int wave = tid >> 5;
  const int wm = wave >> 2;
  const int wn = wave & 3;
  const int half = lane >> 4;
  const int mn = lane & 15;

  v8f c[4][2];
  v8f vzero = {};
#pragma unroll
  for (int mi = 0; mi < 4; ++mi)
#pragma unroll
    for (int ni = 0; ni < 2; ++ni) c[mi][ni] = vzero;

  const unsigned short* aBase = hb + (size_t)(tm * BM) * K;
  const unsigned short* bBase = dwb + (size_t)(tn * BN) * K;
  const int nChunks = K / BK;

  if (wave == 0) {
    tdm_load_tile((0 * TILE_USH) * 2, aBase, K);
    tdm_load_tile((2 * TILE_USH) * 2, bBase, K);
  }

  for (int i = 0; i < nChunks; ++i) {
    const int buf = i & 1;
    const int aOff = buf * TILE_USH;
    const int bOff = (2 + buf) * TILE_USH;
    if (wave == 0) __builtin_amdgcn_s_wait_tensorcnt(0);
    __syncthreads();
    if (i + 1 < nChunks && wave == 0) {
      const int kn = (i + 1) * BK;
      tdm_load_tile((buf ^ 1) * TILE_USH * 2, aBase + kn, K);
      tdm_load_tile((2 + (buf ^ 1)) * TILE_USH * 2, bBase + kn, K);
    }

#pragma unroll
    for (int ks = 0; ks < BK; ks += 32) {
      Frag a[4], b[2];
#pragma unroll
      for (int mi = 0; mi < 4; ++mi) {
        int base = aOff + (wm * 64 + mi * 16 + mn) * LDK + ks + half * 8;
        a[mi].q[0] = *(const uint4*)&smem[base];
        a[mi].q[1] = *(const uint4*)&smem[base + 16];
      }
#pragma unroll
      for (int ni = 0; ni < 2; ++ni) {
        int base = bOff + (wn * 32 + ni * 16 + mn) * LDK + ks + half * 8;
        b[ni].q[0] = *(const uint4*)&smem[base];
        b[ni].q[1] = *(const uint4*)&smem[base + 16];
      }
#pragma unroll
      for (int mi = 0; mi < 4; ++mi)
#pragma unroll
        for (int ni = 0; ni < 2; ++ni)
          c[mi][ni] = __builtin_amdgcn_wmma_f32_16x16x32_bf16(
              false, a[mi].v, false, b[ni].v, (short)0, c[mi][ni], false, false);
    }
    __syncthreads();   // compute done before next TDM overwrites this buffer pair
  }

#pragma unroll
  for (int mi = 0; mi < 4; ++mi)
#pragma unroll
    for (int ni = 0; ni < 2; ++ni)
#pragma unroll
      for (int r = 0; r < 8; ++r) {
        int p = tm * BM + wm * 64 + mi * 16 + half * 8 + r;
        if (p < count) {
          int tok = idx ? idx[p] : p;
          int col = tn * BN + wn * 32 + ni * 16 + mn;
          float ce = cePtr[(size_t)tok * ceStride];
          float val = c[mi][ni][r] * ce;
          size_t o = (size_t)tok * HDIM + col;
          if (beta) out[o] += val; else out[o] = val;
        }
      }
}

// ---------------- host launch ----------------
extern "C" void kernel_launch(void* const* d_in, const int* in_sizes, int n_in,
                              void* d_out, int out_size, void* d_ws, size_t ws_size,
                              hipStream_t stream) {
  (void)in_sizes; (void)n_in; (void)out_size; (void)ws_size;
  const float* x   = (const float*)d_in[0];
  const float* rw  = (const float*)d_in[1];
  const float* gw  = (const float*)d_in[2];
  const float* uw  = (const float*)d_in[3];
  const float* dww = (const float*)d_in[4];
  const float* sgw = (const float*)d_in[5];
  const float* suw = (const float*)d_in[6];
  const float* sdw = (const float*)d_in[7];
  const float* shg = (const float*)d_in[8];
  float* out = (float*)d_out;

  char* ws = (char*)d_ws;
  unsigned short* xb   = (unsigned short*)ws;                 // T*H bf16      33.5 MB
  float* combine       = (float*)(ws + 33554432);             // T*16 f32
  float* sgate         = (float*)(ws + 34078720);             // T f32
  int*   counts        = (int*)(ws + 34111488);               // 17 ints
  int*   idxlist       = (int*)(ws + 34112000);               // 16*T ints
  unsigned short* hbuf = (unsigned short*)(ws + 34636288);    // 8192*5632 bf16 92.3 MB
  unsigned short* wst  = (unsigned short*)(ws + 126910976);   // staged weights 69.2 MB

  float* logits_out = out + (size_t)T_TOK * HDIM;

  const size_t IH  = (size_t)IDIM * HDIM;    // 2,883,584
  const size_t SIH = (size_t)SIDIM * HDIM;   // 11,534,336
  const int IH_BLK  = (int)(IH / (8 * 256));   // 1408
  const int SIH_BLK = (int)(SIH / (8 * 256));  // 5632

  cast_bf16_kernel<<<8192, 256, 0, stream>>>(x, xb);
  router_kernel<<<T_TOK / 8, 256, 0, stream>>>(x, rw, shg, logits_out, combine, sgate);
  dispatch_kernel<<<NEXP, 256, 0, stream>>>(combine, counts, idxlist);

  // ---- shared expert (dense; beta=0 initializes out) ----
  cast_bf16_kernel<<<SIH_BLK, 256, 0, stream>>>(sgw, wst);
  cast_bf16_kernel<<<SIH_BLK, 256, 0, stream>>>(suw, wst + SIH);
  cast_bf16_kernel<<<SIH_BLK, 256, 0, stream>>>(sdw, wst + 2 * SIH);
  moe_gateup_kernel<<<dim3(T_TOK / BM, SIDIM / BN), 256, 0, stream>>>(
      xb, wst, wst + SIH, hbuf, nullptr, counts + NEXP, HDIM, SIDIM);
  moe_down_kernel<<<dim3(T_TOK / BM, HDIM / BN), 256, 0, stream>>>(
      hbuf, wst + 2 * SIH, sgate, 1, nullptr, counts + NEXP, out, SIDIM, 0);

  // ---- routed experts (sparse, compacted token lists) ----
  for (int e = 0; e < NEXP; ++e) {
    cast_bf16_kernel<<<IH_BLK, 256, 0, stream>>>(gw + e * IH, wst);
    cast_bf16_kernel<<<IH_BLK, 256, 0, stream>>>(uw + e * IH, wst + IH);
    cast_bf16_kernel<<<IH_BLK, 256, 0, stream>>>(dww + e * IH, wst + 2 * IH);
    moe_gateup_kernel<<<dim3(T_TOK / BM, IDIM / BN), 256, 0, stream>>>(
        xb, wst, wst + IH, hbuf, idxlist + e * T_TOK, counts + e, HDIM, IDIM);
    moe_down_kernel<<<dim3(T_TOK / BM, HDIM / BN), 256, 0, stream>>>(
        hbuf, wst + 2 * IH, combine + e, NEXP, idxlist + e * T_TOK, counts + e,
        out, IDIM, 1);
  }
}
